// GNNReGVD_48112223650342
// MI455X (gfx1250) — matrix operand order repaired and checked by reading, not compiled, not run on hardware
//
#include <hip/hip_runtime.h>
#include <stdint.h>

typedef __bf16 bf16;
typedef __attribute__((ext_vector_type(16))) __bf16 bf16x16;
typedef __attribute__((ext_vector_type(8)))  float  f32x8;

union ABFrag { bf16x16 v; uint4 q[2]; };

__device__ __forceinline__ f32x8 wmma_bf16(bf16x16 a, bf16x16 b, f32x8 c) {
  // D = A(16x32 bf16) * B(32x16 bf16) + C(16x16 f32)
  return __builtin_amdgcn_wmma_f32_16x16x32_bf16(false, a, false, b, (short)0, c, false, false);
}

// ---------------- elementwise f32 -> bf16 ----------------
__global__ void f2bf_kernel(const float* __restrict__ in, bf16* __restrict__ out, int n) {
  int i = blockIdx.x * blockDim.x + threadIdx.x;
  if (i < n) out[i] = (bf16)in[i];
}

// ---------------- [Z,R,C] f32 -> [Z,C,R] bf16 ----------------
__global__ __launch_bounds__(256) void tf2bf_kernel(const float* __restrict__ in,
                                                    bf16* __restrict__ out, int R, int C) {
  __shared__ float t[32][33];
  const int z = blockIdx.z;
  const int c0 = blockIdx.x * 32, r0 = blockIdx.y * 32;
  const float* ip = in + (size_t)z * R * C;
  bf16* op = out + (size_t)z * R * C;
#pragma unroll
  for (int k = 0; k < 4; ++k)
    t[threadIdx.y + 8 * k][threadIdx.x] =
        ip[(size_t)(r0 + threadIdx.y + 8 * k) * C + c0 + threadIdx.x];
  __syncthreads();
#pragma unroll
  for (int k = 0; k < 4; ++k)
    op[(size_t)(c0 + threadIdx.y + 8 * k) * R + r0 + threadIdx.x] =
        (bf16)t[threadIdx.x][threadIdx.y + 8 * k];
}

// ---------------- [Z,R,C] bf16 -> [Z,C,R] bf16 ----------------
__global__ __launch_bounds__(256) void tbf_kernel(const bf16* __restrict__ in,
                                                  bf16* __restrict__ out, int R, int C) {
  __shared__ bf16 t[32][34];
  const int z = blockIdx.z;
  const int c0 = blockIdx.x * 32, r0 = blockIdx.y * 32;
  const bf16* ip = in + (size_t)z * R * C;
  bf16* op = out + (size_t)z * R * C;
#pragma unroll
  for (int k = 0; k < 4; ++k)
    t[threadIdx.y + 8 * k][threadIdx.x] =
        ip[(size_t)(r0 + threadIdx.y + 8 * k) * C + c0 + threadIdx.x];
  __syncthreads();
#pragma unroll
  for (int k = 0; k < 4; ++k)
    op[(size_t)(c0 + threadIdx.y + 8 * k) * R + r0 + threadIdx.x] =
        t[threadIdx.x][threadIdx.y + 8 * k];
}

// ---------------- tiled bf16 WMMA GEMM (LDS ping-pong): C = A[M,K] * Bt[N,K]^T ----------------
// EPI: 0 = store bf16, 1 = f32 + bias, 2 = f32 + bias + relu
template <int EPI>
__global__ __launch_bounds__(128) void gemm_bf16_kernel(
    const bf16* __restrict__ A, const bf16* __restrict__ Bt,
    void* __restrict__ Cv, const float* __restrict__ bias,
    int M, int Nn, int K,
    int zA_div, long long zA_str, int zB_mod, long long zB_str, long long zC_str) {
  __shared__ bf16 As[2][128 * 40];
  __shared__ bf16 Bs[2][128 * 40];
  const int z = blockIdx.z;
  A  += (size_t)(z / zA_div) * zA_str;
  Bt += (size_t)(z % zB_mod) * zB_str;
  const int m0 = blockIdx.x * 128, n0 = blockIdx.y * 128;
  const int tid = threadIdx.x, lane = tid & 31, wid = tid >> 5;
  const int wm = (wid >> 1) * 64, wn = (wid & 1) * 64;
  const int half = lane >> 4, l15 = lane & 15;

  const bf16* Abase = A + (size_t)(m0 + tid) * K;
  const bf16* Bbase = Bt + (size_t)(n0 + tid) * K;

  uint4 ra[4], rb[4];
  auto gload = [&](int k0) {
    const uint4* ga = (const uint4*)(Abase + k0);
    ra[0] = ga[0]; ra[1] = ga[1]; ra[2] = ga[2]; ra[3] = ga[3];
    const uint4* gb = (const uint4*)(Bbase + k0);
    rb[0] = gb[0]; rb[1] = gb[1]; rb[2] = gb[2]; rb[3] = gb[3];
  };
  auto sstore = [&](int buf) {
    uint4* da = (uint4*)&As[buf][tid * 40];
    da[0] = ra[0]; da[1] = ra[1]; da[2] = ra[2]; da[3] = ra[3];
    uint4* db = (uint4*)&Bs[buf][tid * 40];
    db[0] = rb[0]; db[1] = rb[1]; db[2] = rb[2]; db[3] = rb[3];
  };

  const f32x8 vzero = {0.f, 0.f, 0.f, 0.f, 0.f, 0.f, 0.f, 0.f};
  f32x8 acc[4][4];
#pragma unroll
  for (int i = 0; i < 4; ++i)
#pragma unroll
    for (int j = 0; j < 4; ++j) acc[i][j] = vzero;

  const int nk = K >> 5;
  gload(0);
  sstore(0);
  __syncthreads();

  for (int s = 0; s < nk; ++s) {
    const int cur = s & 1;
    if (s + 1 < nk) gload((s + 1) * 32);
    if (s + 2 < nk) {  // global_prefetch_b8 the tile after next
      __builtin_prefetch(Abase + (s + 2) * 32, 0, 1);
      __builtin_prefetch(Bbase + (s + 2) * 32, 0, 1);
    }

    ABFrag a[4], b[4];
#pragma unroll
    for (int i = 0; i < 4; ++i) {
      const uint4* p = (const uint4*)&As[cur][(wm + 16 * i + l15) * 40];
      a[i].q[0] = p[half];          // K = half*8 .. +7
      a[i].q[1] = p[2 + half];      // K = 16 + half*8 .. +7
    }
#pragma unroll
    for (int j = 0; j < 4; ++j) {
      const uint4* p = (const uint4*)&Bs[cur][(wn + 16 * j + l15) * 40];
      b[j].q[0] = p[half];          // K = half*8 .. +7
      b[j].q[1] = p[2 + half];      // K = 16 + half*8 .. +7
    }
#pragma unroll
    for (int i = 0; i < 4; ++i)
#pragma unroll
      for (int j = 0; j < 4; ++j) acc[i][j] = wmma_bf16(a[i].v, b[j].v, acc[i][j]);

    if (s + 1 < nk) sstore((s + 1) & 1);
    __syncthreads();
  }

  bf16*  Cb = (bf16*)Cv  + (size_t)z * zC_str;
  float* Cf = (float*)Cv + (size_t)z * zC_str;
#pragma unroll
  for (int i = 0; i < 4; ++i) {
#pragma unroll
    for (int j = 0; j < 4; ++j) {
      const int col = n0 + wn + 16 * j + l15;
      const int rbase = m0 + wm + 16 * i + half * 8;
      const float bv = (EPI != 0) ? bias[col] : 0.f;
#pragma unroll
      for (int r = 0; r < 8; ++r) {
        float v = acc[i][j][r];
        if (EPI == 0) {
          Cb[(size_t)(rbase + r) * Nn + col] = (bf16)v;
        } else {
          v += bv;
          if (EPI == 2) v = fmaxf(v, 0.f);
          Cf[(size_t)(rbase + r) * Nn + col] = v;
        }
      }
    }
  }
}

// ---------------- per-row dual dot: s1 = h . a[:O], s2 = h . a[O:2O] ----------------
__global__ __launch_bounds__(256) void sdot_kernel(const bf16* __restrict__ h,
                                                   const float* __restrict__ a,
                                                   float* __restrict__ s1, float* __restrict__ s2,
                                                   int Nn, int O, int H) {
  const int z = blockIdx.y;
  const int head = z % H;
  const int row = blockIdx.x * 8 + (threadIdx.x >> 5);
  const int lane = threadIdx.x & 31;
  const bf16* hr = h + ((size_t)z * Nn + row) * O + lane * 8;
  const float* a1 = a + (size_t)head * 2 * O + lane * 8;
  const float* a2 = a1 + O;
  float d1 = 0.f, d2 = 0.f;
#pragma unroll
  for (int e = 0; e < 8; ++e) {
    float f = (float)hr[e];
    d1 += f * a1[e];
    d2 += f * a2[e];
  }
#pragma unroll
  for (int s = 16; s >= 1; s >>= 1) {
    d1 += __shfl_xor(d1, s, 32);
    d2 += __shfl_xor(d2, s, 32);
  }
  if (lane == 0) {
    s1[(size_t)z * Nn + row] = d1;
    s2[(size_t)z * Nn + row] = d2;
  }
}

// ---------------- flash-style masked GAT attention: out = softmax_mask(e) @ h ----------------
#define AN 2048
#define AO 256

// unnormalized prob element (exp(e - m) <= 1, safe in bf16); row-sum l applied at epilogue
#define PBUILD(IDX, AV, SV)                              \
  {                                                      \
    float pe_ = 0.f;                                     \
    if ((AV) > 0.f) {                                    \
      float e_ = s1v + (SV);                             \
      e_ = e_ > 0.f ? e_ : 0.2f * e_;                    \
      pe_ = __expf(e_ - m);                              \
    }                                                    \
    lsum += pe_;                                         \
    p.v[IDX] = (bf16)pe_;                                \
  }

__global__ __launch_bounds__(128) void attn_kernel(
    const float* __restrict__ adj,   // [B,N,N]
    const float* __restrict__ s1g,   // [Z,N]
    const float* __restrict__ s2g,   // [Z,N]
    const bf16*  __restrict__ hT,    // [Z,O,N]  (h transposed)
    const float* __restrict__ basef, // [B,N,O] residual (nullable)
    bf16* __restrict__ outb,         // [B,N,out_ld]
    int H, int out_ld, int do_elu) {
  __shared__ __align__(16) float s2s[AN];   // 8 KB
  __shared__ bf16 hTs[2][AO * 40];          // 2 x 20 KB ping-pong: hT[:, j0:j0+32]
  const int z = blockIdx.y;
  const int b = z / H, head = z % H;
  const int tid = threadIdx.x, lane = tid & 31, wid = tid >> 5;
  const int half = lane >> 4, l15 = lane & 15;
  const int iw = blockIdx.x * 64 + wid * 16;  // 16 rows per wave

  for (int j = tid; j < AN; j += 128) s2s[j] = s2g[(size_t)z * AN + j];
  __syncthreads();

  const int gi = iw + l15;
  const float s1v = s1g[(size_t)z * AN + gi];
  const float* adjrow = adj + ((size_t)b * AN + gi) * AN;

  // phase 1: masked row max only (float4 loads, no exp, no serial chain)
  float m = -3.0e38f;
  for (int jb = half * 4; jb < AN; jb += 8) {
    float4 av = *(const float4*)(adjrow + jb);
    float4 sv = *(const float4*)(s2s + jb);
    if (av.x > 0.f) { float e = s1v + sv.x; e = e > 0.f ? e : 0.2f * e; m = fmaxf(m, e); }
    if (av.y > 0.f) { float e = s1v + sv.y; e = e > 0.f ? e : 0.2f * e; m = fmaxf(m, e); }
    if (av.z > 0.f) { float e = s1v + sv.z; e = e > 0.f ? e : 0.2f * e; m = fmaxf(m, e); }
    if (av.w > 0.f) { float e = s1v + sv.w; e = e > 0.f ? e : 0.2f * e; m = fmaxf(m, e); }
  }
  m = fmaxf(m, __shfl_xor(m, 16, 32));

  // staging: hT tile global -> regs -> LDS (ping-pong)
  const bf16* hTbase = hT + (size_t)z * AO * AN;
  uint4 rh[8];
  auto gload_h = [&](int j0) {
    const uint4* s0 = (const uint4*)(hTbase + (size_t)(tid * 2) * AN + j0);
    rh[0] = s0[0]; rh[1] = s0[1]; rh[2] = s0[2]; rh[3] = s0[3];
    const uint4* s1p = (const uint4*)(hTbase + (size_t)(tid * 2 + 1) * AN + j0);
    rh[4] = s1p[0]; rh[5] = s1p[1]; rh[6] = s1p[2]; rh[7] = s1p[3];
  };
  auto sstore_h = [&](int buf) {
    uint4* d0 = (uint4*)&hTs[buf][(tid * 2) * 40];
    d0[0] = rh[0]; d0[1] = rh[1]; d0[2] = rh[2]; d0[3] = rh[3];
    uint4* d1 = (uint4*)&hTs[buf][(tid * 2 + 1) * 40];
    d1[0] = rh[4]; d1[1] = rh[5]; d1[2] = rh[6]; d1[3] = rh[7];
  };

  const f32x8 vzero = {0.f, 0.f, 0.f, 0.f, 0.f, 0.f, 0.f, 0.f};
  f32x8 acc[16];
#pragma unroll
  for (int t = 0; t < 16; ++t) acc[t] = vzero;

  gload_h(0);
  sstore_h(0);
  __syncthreads();

  float lsum = 0.f;
  const int nt = AN / 32;
  for (int s = 0; s < nt; ++s) {
    const int cur = s & 1;
    const int j0 = s * 32;
    if (s + 1 < nt) gload_h(j0 + 32);

    // build un-normalized P tile directly in the WMMA A-fragment layout (float4 loads)
    ABFrag p;
    {
      const int ja = j0 + half * 16;
      const int jb2 = j0 + half * 16 + 8;
      float4 a0 = *(const float4*)(adjrow + ja);
      float4 a1 = *(const float4*)(adjrow + ja + 4);
      float4 a2 = *(const float4*)(adjrow + jb2);
      float4 a3 = *(const float4*)(adjrow + jb2 + 4);
      float4 q0 = *(const float4*)(s2s + ja);
      float4 q1 = *(const float4*)(s2s + ja + 4);
      float4 q2 = *(const float4*)(s2s + jb2);
      float4 q3 = *(const float4*)(s2s + jb2 + 4);
      PBUILD(0, a0.x, q0.x) PBUILD(1, a0.y, q0.y) PBUILD(2, a0.z, q0.z) PBUILD(3, a0.w, q0.w)
      PBUILD(4, a1.x, q1.x) PBUILD(5, a1.y, q1.y) PBUILD(6, a1.z, q1.z) PBUILD(7, a1.w, q1.w)
      PBUILD(8, a2.x, q2.x) PBUILD(9, a2.y, q2.y) PBUILD(10, a2.z, q2.z) PBUILD(11, a2.w, q2.w)
      PBUILD(12, a3.x, q3.x) PBUILD(13, a3.y, q3.y) PBUILD(14, a3.z, q3.z) PBUILD(15, a3.w, q3.w)
    }

    // 16 WMMAs over the O dimension, B-fragment ds_loads software-pipelined (depth 2)
    ABFrag bfr[2];
    {
      const uint4* pb = (const uint4*)&hTs[cur][(0 * 16 + l15) * 40];
      bfr[0].q[0] = pb[half * 2];
      bfr[0].q[1] = pb[half * 2 + 1];
    }
#pragma unroll
    for (int t = 0; t < 16; ++t) {
      if (t + 1 < 16) {
        const uint4* pb = (const uint4*)&hTs[cur][((t + 1) * 16 + l15) * 40];
        bfr[(t + 1) & 1].q[0] = pb[half * 2];
        bfr[(t + 1) & 1].q[1] = pb[half * 2 + 1];
      }
      acc[t] = wmma_bf16(p.v, bfr[t & 1].v, acc[t]);
    }

    if (s + 1 < nt) sstore_h((s + 1) & 1);
    __syncthreads();
  }

  // row sums -> 1/l, distributed to epilogue rows via shuffles
  const float lt = lsum + __shfl_xor(lsum, 16, 32);
  const float invl = 1.f / lt;  // >= 1 guaranteed by self-loops
  float inv_r[8];
#pragma unroll
  for (int r = 0; r < 8; ++r) inv_r[r] = __shfl(invl, half * 8 + r, 32);

  // epilogue: normalize, optional elu, optional residual add, bf16 store (concat via out_ld/head)
#pragma unroll
  for (int t = 0; t < 16; ++t) {
    const int col = t * 16 + l15;
    const int rbase = iw + half * 8;
#pragma unroll
    for (int r = 0; r < 8; ++r) {
      const int row = rbase + r;
      float v = acc[t][r] * inv_r[r];
      if (do_elu) v = v > 0.f ? v : __expf(v) - 1.f;
      if (basef) v += basef[((size_t)b * AN + row) * AO + col];
      outb[((size_t)b * AN + row) * out_ld + head * AO + col] = (bf16)v;
    }
  }
}

extern "C" void kernel_launch(void* const* d_in, const int* in_sizes, int n_in,
                              void* d_out, int out_size, void* d_ws, size_t ws_size,
                              hipStream_t stream) {
  const float* x       = (const float*)d_in[0];
  const float* adj     = (const float*)d_in[1];
  // d_in[2] = mask (unused by the reference forward)
  const float* W_heads = (const float*)d_in[3];
  const float* a_heads = (const float*)d_in[4];
  const float* W_out   = (const float*)d_in[5];
  const float* a_out   = (const float*)d_in[6];
  const float* W_lin   = (const float*)d_in[7];
  const float* b_lin   = (const float*)d_in[8];
  const float* W_ln    = (const float*)d_in[9];
  const float* b_ln    = (const float*)d_in[10];
  float* out = (float*)d_out;
  (void)in_sizes; (void)n_in; (void)out_size; (void)ws_size;

  const int B = 2, N = 2048, F = 768, H = 8, O = 256, HO = H * O;
  const int Z0 = B * H;

  char* ws = (char*)d_ws;
  size_t off = 0;
  auto alloc = [&](size_t bytes) -> char* {
    char* p = ws + off;
    off += (bytes + 255) & ~(size_t)255;
    return p;
  };
  bf16*  x_bf    = (bf16*) alloc((size_t)B * N * F * 2);
  bf16*  WheadsT = (bf16*) alloc((size_t)H * O * F * 2);   // [H,O,F] = W^T per head
  bf16*  Wlin_bf = (bf16*) alloc((size_t)O * HO * 2);      // already [N,K] layout
  bf16*  WoutT   = (bf16*) alloc((size_t)O * HO * 2);      // [O,HO] = W_out^T
  bf16*  Wln_bf  = (bf16*) alloc((size_t)O * O * 2);       // already [N,K] layout
  bf16*  h_bf    = (bf16*) alloc((size_t)Z0 * N * O * 2);
  bf16*  hT_bf   = (bf16*) alloc((size_t)Z0 * O * N * 2);
  float* s1      = (float*)alloc((size_t)Z0 * N * 4);
  float* s2      = (float*)alloc((size_t)Z0 * N * 4);
  bf16*  x1_bf   = (bf16*) alloc((size_t)B * N * HO * 2);
  float* lin_f   = (float*)alloc((size_t)B * N * O * 4);
  bf16*  h2_bf   = (bf16*) alloc((size_t)B * N * O * 2);
  bf16*  h2T_bf  = (bf16*) alloc((size_t)B * O * N * 2);
  float* s1b     = (float*)alloc((size_t)B * N * 4);
  float* s2b     = (float*)alloc((size_t)B * N * 4);
  bf16*  x2_bf   = (bf16*) alloc((size_t)B * N * O * 2);

  // --- precision conversion / weight transposition ---
  { int n = B * N * F; f2bf_kernel<<<(n + 255) / 256, 256, 0, stream>>>(x, x_bf, n); }
  { int n = O * HO;    f2bf_kernel<<<(n + 255) / 256, 256, 0, stream>>>(W_lin, Wlin_bf, n); }
  { int n = O * O;     f2bf_kernel<<<(n + 255) / 256, 256, 0, stream>>>(W_ln, Wln_bf, n); }
  tf2bf_kernel<<<dim3(O / 32, F / 32, H), dim3(32, 8), 0, stream>>>(W_heads, WheadsT, F, O);
  tf2bf_kernel<<<dim3(O / 32, HO / 32, 1), dim3(32, 8), 0, stream>>>(W_out, WoutT, HO, O);

  // --- layer 0: h[b,h] = x[b] @ W[h] ---
  gemm_bf16_kernel<0><<<dim3(N / 128, O / 128, Z0), 128, 0, stream>>>(
      x_bf, WheadsT, h_bf, nullptr, N, O, F,
      H, (long long)N * F, H, (long long)O * F, (long long)N * O);
  sdot_kernel<<<dim3(N / 8, Z0), 256, 0, stream>>>(h_bf, a_heads, s1, s2, N, O, H);
  tbf_kernel<<<dim3(O / 32, N / 32, Z0), dim3(32, 8), 0, stream>>>(h_bf, hT_bf, N, O);
  attn_kernel<<<dim3(N / 64, Z0), 128, 0, stream>>>(adj, s1, s2, hT_bf, nullptr, x1_bf, H, HO, 1);

  // --- layer 1 ---
  gemm_bf16_kernel<1><<<dim3(N / 128, O / 128, B), 128, 0, stream>>>(
      x1_bf, Wlin_bf, lin_f, b_lin, N, O, HO,
      1, (long long)N * HO, 1, 0LL, (long long)N * O);
  gemm_bf16_kernel<0><<<dim3(N / 128, O / 128, B), 128, 0, stream>>>(
      x1_bf, WoutT, h2_bf, nullptr, N, O, HO,
      1, (long long)N * HO, 1, 0LL, (long long)N * O);
  sdot_kernel<<<dim3(N / 8, B), 256, 0, stream>>>(h2_bf, a_out, s1b, s2b, N, O, 1);
  tbf_kernel<<<dim3(O / 32, N / 32, B), dim3(32, 8), 0, stream>>>(h2_bf, h2T_bf, N, O);
  attn_kernel<<<dim3(N / 64, B), 128, 0, stream>>>(adj, s1b, s2b, h2T_bf, lin_f, x2_bf, 1, O, 0);

  // --- final: out = relu(x2 @ W_ln^T + b_ln) ---
  gemm_bf16_kernel<2><<<dim3(N / 128, O / 128, B), 128, 0, stream>>>(
      x2_bf, Wln_bf, out, b_ln, N, O, O,
      1, (long long)N * O, 1, 0LL, (long long)N * O);
}
